// TextRNNLSTMRepresentation_74062416052784
// MI455X (gfx1250) — matrix operand order repaired and checked by reading, not compiled, not run on hardware
//
#include <hip/hip_runtime.h>
#include <hip/hip_bf16.h>

// Problem constants
#define BB   256   // batch
#define SS   512   // seq len
#define VV   800   // vocab
#define EE   128   // embed dim
#define HH   256   // hidden
#define G4   1024  // 4*H

// Feature toggles (flip to 0 if a toolchain lane rejects the async/TDM forms)
#define USE_ASYNC_LDS 1
#define USE_TDM_PRELOAD 1

typedef __attribute__((ext_vector_type(16))) _Float16 v16h;
typedef __attribute__((ext_vector_type(8)))  _Float16 v8h;
typedef __attribute__((ext_vector_type(8)))  float    v8f;
typedef __attribute__((ext_vector_type(4)))  unsigned u32x4;
typedef __attribute__((ext_vector_type(4)))  int      i32x4;
typedef __attribute__((ext_vector_type(8)))  int      i32x8;

__device__ __forceinline__ v16h cat8(v8h lo, v8h hi) {
    return __builtin_shufflevector(lo, hi, 0,1,2,3,4,5,6,7,8,9,10,11,12,13,14,15);
}

__device__ __forceinline__ float sigmoidf_(float x) {
    return 1.0f / (1.0f + __expf(-x));
}

// ---------------------------------------------------------------------------
// Prep kernels
// ---------------------------------------------------------------------------
__global__ void cvt_f32_to_f16_kernel(const float* __restrict__ src,
                                      _Float16* __restrict__ dst, int n) {
    int i = blockIdx.x * 256 + threadIdx.x;
    if (i < n) dst[i] = (_Float16)src[i];
}

__global__ void bias_combine_kernel(const float* __restrict__ a,
                                    const float* __restrict__ b,
                                    float* __restrict__ o, int n) {
    int i = blockIdx.x * 256 + threadIdx.x;
    if (i < n) o[i] = a[i] + b[i];
}

// Embedding gather: eh[s*B + b][d] = (idx==0 ? 0 : emb[idx][d]) as f16.
// Output layout (S,B,E) so rows match xp rows (s,b).
__global__ void embed_kernel(const int* __restrict__ x,
                             const float* __restrict__ emb,
                             _Float16* __restrict__ eh) {
    int r = blockIdx.x;              // r = s*B + b
    int s = r >> 8;                  // B == 256
    int b = r & 255;
    int d = threadIdx.x;             // 128 threads == E
    int idx = x[(size_t)b * SS + s];
    float v = (idx == 0) ? 0.0f : emb[(size_t)idx * EE + d];
    eh[(size_t)r * EE + d] = (_Float16)v;
}

// ---------------------------------------------------------------------------
// Input-projection GEMM: xp[r][g] = sum_k A[r][k]*W[g][k] + bias[g]
// A: (M=S*B, K) f16 row-major.  W: (1024, K) f16 row-major.  xp: (M,1024) f32.
// WG = 128 threads = 4 waves. WG tile 64(M) x 64(N); wave w owns n-strip w*16,
// 4 M-tiles of 16 -> 4 WMMA accumulators per wave, K-loop step 32.
// ---------------------------------------------------------------------------
__global__ void __launch_bounds__(128)
gemm_xp_kernel(const _Float16* __restrict__ A,
               const _Float16* __restrict__ W,
               const float* __restrict__ bias,
               float* __restrict__ xp, int K) {
    const int lane = threadIdx.x & 31;
    const int wv   = threadIdx.x >> 5;
    const int n0   = blockIdx.x * 64 + wv * 16;
    const long m0  = (long)blockIdx.y * 64;
    const int ln   = lane & 15;
    const int klo  = (lane & 16) ? 8 : 0;    // A fragment K base (halves)
    const int koff = (lane & 16) ? 16 : 0;   // unused; B fragment uses klo like A

    v8f acc[4] = {};
    const _Float16* wrow = W + (size_t)(n0 + ln) * K;

    for (int k0 = 0; k0 < K; k0 += 32) {
        v16h bf = cat8(*(const v8h*)(wrow + k0 + klo),
                       *(const v8h*)(wrow + k0 + klo + 16));
#pragma unroll
        for (int mt = 0; mt < 4; ++mt) {
            const _Float16* ar = A + (size_t)(m0 + mt * 16 + ln) * K + k0 + klo;
            v16h af = cat8(*(const v8h*)ar, *(const v8h*)(ar + 16));
            acc[mt] = __builtin_amdgcn_wmma_f32_16x16x32_f16(
                false, af, false, bf, (short)0, acc[mt], false, false);
        }
    }

    const float bv = bias[n0 + ln];
    const int mhi = (lane >> 4) << 3;        // 0 or 8
#pragma unroll
    for (int mt = 0; mt < 4; ++mt) {
        long mb = m0 + mt * 16 + mhi;
#pragma unroll
        for (int v = 0; v < 8; ++v)
            xp[(size_t)(mb + v) * G4 + n0 + ln] = acc[mt][v] + bv;
    }
}

// ---------------------------------------------------------------------------
// Persistent LSTM scan. 128 WGs x 128 threads (4 waves), device-wide barrier.
// WG owns patch b0..b0+31 (rows) x j0..j0+15 (hidden cols), i.e. gate columns
// {j0..j0+15} of each of the 4 gate quarters. Wave w computes gate quarter w.
// Whh slice (4 x 16 x 256 f16 = 32KB) preloaded to LDS once (TDM tensor load
// when available); h (32x256 f16 = 16KB) staged per step with async global->
// LDS loads (ASYNCcnt); gates exchanged via 8KB LDS; cell state c in VGPRs.
// ---------------------------------------------------------------------------
#define NWG_SCAN 128

__global__ void __launch_bounds__(128)
lstm_scan_kernel(const float* __restrict__ xp,       // (S,B,4H) f32
                 const _Float16* __restrict__ WhhH,  // (4H,H) f16
                 _Float16* __restrict__ hbuf0,       // (B,H) f16  (zeroed)
                 _Float16* __restrict__ hbuf1,       // (B,H) f16  (zeroed)
                 _Float16* __restrict__ hseq,        // (S,B,H) f16 or null
                 float* __restrict__ hn,             // (B,H) f32 or null
                 unsigned* __restrict__ syncv) {
    __shared__ __align__(16) _Float16 whh_s[4][16][HH];  // 32 KB
    __shared__ __align__(16) _Float16 h_s[32][HH];       // 16 KB
    __shared__ float gx[4][32][16];                      //  8 KB

    const int tid  = threadIdx.x;
    const int lane = tid & 31;
    const int wv   = tid >> 5;                 // gate quarter 0..3
    const int b0   = (blockIdx.x >> 4) * 32;   // 8 b-tiles
    const int j0   = (blockIdx.x & 15) * 16;   // 16 j-tiles
    const int ln   = lane & 15;
    const int klo  = (lane & 16) ? 8 : 0;
    const int koff = (lane & 16) ? 16 : 0;
    const int mhi  = (lane >> 4) << 3;

    // ---- Preload Whh slice: quarter q rows [q*256 + j0, +16) (contiguous
    //      16*256 f16 = 8 KB chunk per quarter). Wave wv moves quarter wv.
#if USE_TDM_PRELOAD && __has_builtin(__builtin_amdgcn_tensor_load_to_lds) && \
    __has_builtin(__builtin_amdgcn_s_wait_tensorcnt)
    {
        // Tensor DMA descriptor (D#), groups per CDNA5 ISA.
        const uint64_t ga =
            (uint64_t)(uintptr_t)(WhhH + ((size_t)wv * HH + j0) * HH);
        const unsigned lds_addr = (unsigned)(uintptr_t)&whh_s[wv][0][0];
        u32x4 g0;
        g0[0] = 1u;                                   // count=1, user mode
        g0[1] = lds_addr;                             // lds_addr
        g0[2] = (unsigned)ga;                         // global_addr[31:0]
        g0[3] = (unsigned)((ga >> 32) & 0x1FFFFFFu)   // global_addr[56:32]
              | 0x80000000u;                          // type=2 ("image")
        i32x8 g1;
        g1[0] = (int)(1u << 16);          // workgroup_mask=0, data_size=2B
        g1[1] = (int)(0x2000u << 16);     // tensor_dim0 = 8192 elems (lo16)
        g1[2] = (int)(1u << 16);          // tensor_dim0 hi=0 | tensor_dim1=1
        g1[3] = (int)(0x2000u << 16);     // tile_dim0 = 8192 elems
        g1[4] = 1;                        // tile_dim1 = 1 (tile_dim2 = 0)
        g1[5] = 8192;                     // tensor_dim0_stride = 8192
        g1[6] = 0;
        g1[7] = 0;
        i32x4 gz4 = {0, 0, 0, 0};         // groups 2/3 unused (<=2D tensor)
        i32x8 gz8 = {0, 0, 0, 0, 0, 0, 0, 0};
        __builtin_amdgcn_tensor_load_to_lds(g0, g1, gz4, gz4, gz8, 0);
        __builtin_amdgcn_s_wait_tensorcnt(0);
    }
#else
    {
        uint4* dst = (uint4*)&whh_s[0][0][0];
        for (int idx = tid; idx < 2048; idx += 128) {       // 2048 uint4 total
            int q = idx >> 9;                               // 512 uint4/quarter
            int r = idx & 511;
            const uint4* s =
                (const uint4*)(WhhH + ((size_t)q * HH + j0) * HH);
            dst[idx] = s[r];
        }
    }
#endif
    float c[4] = {0.f, 0.f, 0.f, 0.f};
    _Float16* hc = hbuf0;
    _Float16* hx = hbuf1;
    __syncthreads();

    for (int t = 0; t < SS; ++t) {
        // ---- Stage h_t rows b0..b0+31 (contiguous 16 KB) into LDS.
#if USE_ASYNC_LDS
        {
            const uint64_t gbase = (uint64_t)(uintptr_t)(hc + (size_t)b0 * HH);
            const unsigned lbase = (unsigned)(uintptr_t)&h_s[0][0];
            for (int idx = tid; idx < 1024; idx += 128) {   // 8 issues / lane
                unsigned la = lbase + (unsigned)idx * 16u;
                uint64_t gaddr = gbase + (uint64_t)idx * 16u;
                asm volatile("global_load_async_to_lds_b128 %0, %1, off"
                             :: "v"(la), "v"(gaddr) : "memory");
            }
            asm volatile("s_wait_asynccnt 0x0" ::: "memory");
        }
#else
        {
            const uint4* src = (const uint4*)(hc + (size_t)b0 * HH);
            uint4* dst = (uint4*)&h_s[0][0];
            for (int idx = tid; idx < 1024; idx += 128) dst[idx] = src[idx];
        }
#endif
        __syncthreads();

        // gates(32 x 16, quarter wv) = h(32x256) @ Whh_q^T(256x16)
        v8f acc[2] = {};
#pragma unroll
        for (int kk = 0; kk < 8; ++kk) {
            const int k0 = kk * 32;
            v16h bf = cat8(*(const v8h*)&whh_s[wv][ln][k0 + klo],
                           *(const v8h*)&whh_s[wv][ln][k0 + klo + 16]);
#pragma unroll
            for (int mt = 0; mt < 2; ++mt) {
                const _Float16* ar = &h_s[mt * 16 + ln][k0 + klo];
                v16h af = cat8(*(const v8h*)ar, *(const v8h*)(ar + 16));
                acc[mt] = __builtin_amdgcn_wmma_f32_16x16x32_f16(
                    false, af, false, bf, (short)0, acc[mt], false, false);
            }
        }
        // Scatter gate fragments to LDS for cross-wave cell update.
#pragma unroll
        for (int mt = 0; mt < 2; ++mt)
#pragma unroll
            for (int v = 0; v < 8; ++v)
                gx[wv][mt * 16 + mhi + v][ln] = acc[mt][v];
        __syncthreads();

        // Cell update: 512 elements / 128 threads = 4 each.
#pragma unroll
        for (int p = 0; p < 4; ++p) {
            const int linear = tid + p * 128;
            const int bb = linear >> 4;
            const int jj = linear & 15;
            const size_t xb = ((size_t)t * BB + b0 + bb) * G4 + j0 + jj;
            float gi = gx[0][bb][jj] + xp[xb];
            float gf = gx[1][bb][jj] + xp[xb + HH];
            float gg = gx[2][bb][jj] + xp[xb + 2 * HH];
            float go = gx[3][bb][jj] + xp[xb + 3 * HH];
            gi = sigmoidf_(gi);
            gf = sigmoidf_(gf);
            gg = tanhf(gg);
            go = sigmoidf_(go);
            c[p] = gf * c[p] + gi * gg;
            const float hv = go * tanhf(c[p]);
            const size_t hidx = (size_t)(b0 + bb) * HH + j0 + jj;
            hx[hidx] = (_Float16)hv;
            if (hseq) hseq[(size_t)t * (BB * HH) + hidx] = (_Float16)hv;
            if (hn && t == SS - 1) hn[hidx] = hv;
        }
        if (t + 1 < SS)   // warm next step's xp tile into cache
            __builtin_prefetch(
                xp + ((size_t)(t + 1) * BB + b0 + (tid >> 4)) * G4 + j0, 0, 1);

        // Device-wide barrier: all h_{t+1} written before anyone reads it.
        __syncthreads();
        if (tid == 0) {
            __threadfence();
            const unsigned target = (unsigned)(t + 1) * NWG_SCAN;
            __hip_atomic_fetch_add(syncv, 1u, __ATOMIC_ACQ_REL,
                                   __HIP_MEMORY_SCOPE_AGENT);
            while (__hip_atomic_load(syncv, __ATOMIC_ACQUIRE,
                                     __HIP_MEMORY_SCOPE_AGENT) < target) {}
        }
        __syncthreads();
        _Float16* tmp = hc; hc = hx; hx = tmp;
    }
}

// ---------------------------------------------------------------------------
// LayerNorm + projection: one block (128 thr) per batch row.
// ---------------------------------------------------------------------------
__global__ void __launch_bounds__(128)
final_kernel(const float* __restrict__ hn, const float* __restrict__ g,
             const float* __restrict__ be, const float* __restrict__ pW,
             const float* __restrict__ pb, float* __restrict__ out) {
    __shared__ float row[HH];
    __shared__ float red[128];
    const int b = blockIdx.x, tid = threadIdx.x;
    const float v0 = hn[(size_t)b * HH + tid];
    const float v1 = hn[(size_t)b * HH + 128 + tid];

    red[tid] = v0 + v1;
    __syncthreads();
    for (int off = 64; off > 0; off >>= 1) {
        if (tid < off) red[tid] += red[tid + off];
        __syncthreads();
    }
    const float mu = red[0] / 256.0f;
    __syncthreads();
    const float d0 = v0 - mu, d1 = v1 - mu;
    red[tid] = d0 * d0 + d1 * d1;
    __syncthreads();
    for (int off = 64; off > 0; off >>= 1) {
        if (tid < off) red[tid] += red[tid + off];
        __syncthreads();
    }
    const float rstd = rsqrtf(red[0] / 256.0f + 1e-5f);
    __syncthreads();
    row[tid]       = d0 * rstd * g[tid]       + be[tid];
    row[tid + 128] = d1 * rstd * g[tid + 128] + be[tid + 128];
    __syncthreads();

    const float* w = pW + (size_t)tid * HH;   // proj_W row e = tid
    float acc = pb[tid];
    for (int j = 0; j < HH; ++j) acc += row[j] * w[j];
    out[(size_t)b * EE + tid] = acc;
}

// ---------------------------------------------------------------------------
// Host-side orchestration (graph-capture safe: kernels + hipMemsetAsync only)
// ---------------------------------------------------------------------------
extern "C" void kernel_launch(void* const* d_in, const int* in_sizes, int n_in,
                              void* d_out, int out_size, void* d_ws, size_t ws_size,
                              hipStream_t stream) {
    const int*   x     = (const int*)  d_in[0];
    const float* emb   = (const float*)d_in[1];
    const float* Wih0  = (const float*)d_in[2];
    const float* Whh0  = (const float*)d_in[3];
    const float* bih0  = (const float*)d_in[4];
    const float* bhh0  = (const float*)d_in[5];
    const float* Wih1  = (const float*)d_in[6];
    const float* Whh1  = (const float*)d_in[7];
    const float* bih1  = (const float*)d_in[8];
    const float* bhh1  = (const float*)d_in[9];
    const float* ln_g  = (const float*)d_in[10];
    const float* ln_b  = (const float*)d_in[11];
    const float* projW = (const float*)d_in[12];
    const float* projb = (const float*)d_in[13];
    float* out = (float*)d_out;

    // Workspace carving (256B aligned)
    char* w = (char*)d_ws;
    auto alloc = [&](size_t bytes) -> char* {
        char* p = w;
        w += (bytes + 255) & ~(size_t)255;
        return p;
    };
    _Float16* eh    = (_Float16*)alloc((size_t)SS * BB * EE * 2);   //  32 MB
    _Float16* h1seq = (_Float16*)alloc((size_t)SS * BB * HH * 2);   //  64 MB
    float*    xp    = (float*)   alloc((size_t)SS * BB * G4 * 4);   // 512 MB
    _Float16* Wih0h = (_Float16*)alloc((size_t)G4 * EE * 2);
    _Float16* Whh0h = (_Float16*)alloc((size_t)G4 * HH * 2);
    _Float16* Wih1h = (_Float16*)alloc((size_t)G4 * HH * 2);
    _Float16* Whh1h = (_Float16*)alloc((size_t)G4 * HH * 2);
    float*    bias0 = (float*)   alloc(G4 * 4);
    float*    bias1 = (float*)   alloc(G4 * 4);
    _Float16* hbuf0 = (_Float16*)alloc((size_t)BB * HH * 2);
    _Float16* hbuf1 = (_Float16*)alloc((size_t)BB * HH * 2);
    float*    hn    = (float*)   alloc((size_t)BB * HH * 4);
    unsigned* syncv = (unsigned*)alloc(256);

    // 1) Weight / bias prep (f32 -> f16, bias combine)
    cvt_f32_to_f16_kernel<<<(G4 * EE + 255) / 256, 256, 0, stream>>>(Wih0, Wih0h, G4 * EE);
    cvt_f32_to_f16_kernel<<<(G4 * HH + 255) / 256, 256, 0, stream>>>(Whh0, Whh0h, G4 * HH);
    cvt_f32_to_f16_kernel<<<(G4 * HH + 255) / 256, 256, 0, stream>>>(Wih1, Wih1h, G4 * HH);
    cvt_f32_to_f16_kernel<<<(G4 * HH + 255) / 256, 256, 0, stream>>>(Whh1, Whh1h, G4 * HH);
    bias_combine_kernel<<<(G4 + 255) / 256, 256, 0, stream>>>(bih0, bhh0, bias0, G4);
    bias_combine_kernel<<<(G4 + 255) / 256, 256, 0, stream>>>(bih1, bhh1, bias1, G4);

    // 2) Embedding gather -> (S,B,E) f16
    embed_kernel<<<SS * BB, EE, 0, stream>>>(x, emb, eh);

    // 3) Layer-0 input projection: xp = eh @ Wih0^T + bias0   (WMMA)
    gemm_xp_kernel<<<dim3(G4 / 64, (SS * BB) / 64), 128, 0, stream>>>(
        eh, Wih0h, bias0, xp, EE);

    // 4) Layer-0 persistent scan (WMMA recurrent GEMM + device barrier)
    (void)hipMemsetAsync(syncv, 0, 256, stream);
    (void)hipMemsetAsync(hbuf0, 0, (size_t)BB * HH * 2, stream);
    (void)hipMemsetAsync(hbuf1, 0, (size_t)BB * HH * 2, stream);
    lstm_scan_kernel<<<NWG_SCAN, 128, 0, stream>>>(
        xp, Whh0h, hbuf0, hbuf1, h1seq, nullptr, syncv);

    // 5) Layer-1 input projection: xp = h1seq @ Wih1^T + bias1 (WMMA)
    gemm_xp_kernel<<<dim3(G4 / 64, (SS * BB) / 64), 128, 0, stream>>>(
        h1seq, Wih1h, bias1, xp, HH);

    // 6) Layer-1 persistent scan; capture final h into hn (f32)
    (void)hipMemsetAsync(syncv, 0, 256, stream);
    (void)hipMemsetAsync(hbuf0, 0, (size_t)BB * HH * 2, stream);
    (void)hipMemsetAsync(hbuf1, 0, (size_t)BB * HH * 2, stream);
    lstm_scan_kernel<<<NWG_SCAN, 128, 0, stream>>>(
        xp, Whh1h, hbuf0, hbuf1, nullptr, hn, syncv);

    // 7) LayerNorm + projection
    final_kernel<<<BB, 128, 0, stream>>>(hn, ln_g, ln_b, projW, projb, out);

    (void)in_sizes; (void)n_in; (void)out_size; (void)ws_size;
}
